// TimeSeriesModel_1Step_ES_44968307589364
// MI455X (gfx1250) — compile-verified, matrix operand order checked by reading, not run on hardware
//
#include <hip/hip_runtime.h>
#include <hip/hip_bf16.h>

// CDNA5 / gfx1250: wave32, WMMA 16x16x32 bf16 -> f32 accum.
typedef __attribute__((ext_vector_type(16))) __bf16 v16bf;
typedef __attribute__((ext_vector_type(8)))  float  v8f;

#define BB 1024
#define TT 2048
#define FF 16
#define HH 128
#define CC 2
#define ALPHA 0.3f
#define BETA  0.5f

// ---- fast transcendentals: v_tanh_f32 if available, else v_exp + v_rcp ----
__device__ __forceinline__ float ftanh_(float x) {
#if __has_builtin(__builtin_amdgcn_tanhf)
    return __builtin_amdgcn_tanhf(x);
#else
    // tanh(x) = 1 - 2/(exp(2x)+1)
    return 1.0f - 2.0f * __builtin_amdgcn_rcpf(1.0f + __expf(2.0f * x));
#endif
}
__device__ __forceinline__ float fsigmoid_(float x) {
#if __has_builtin(__builtin_amdgcn_tanhf)
    return __builtin_amdgcn_tanhf(x * 0.5f) * 0.5f + 0.5f;
#else
    return __builtin_amdgcn_rcpf(1.0f + __expf(-x));
#endif
}

// ---- split workgroup barrier (mnemonics verified from gfx1250 disasm) ----
__device__ __forceinline__ void lds_drain_()  { asm volatile("s_wait_dscnt 0x0" ::: "memory"); }
__device__ __forceinline__ void bar_signal_() { asm volatile("s_barrier_signal -1" ::: "memory"); }
__device__ __forceinline__ void bar_wait_()   { asm volatile("s_barrier_wait -1" ::: "memory"); }

// One workgroup = 16 batch rows, 8 waves. Wave w owns hidden cols [16w,16w+16):
// gate tiles r=n_tile w, z=n_tile 8+w, n=n_tile 16+w, so gate math stays in-wave.
// Fragment buffers are ping-ponged so each timestep needs only ONE barrier:
// iteration t reads buf[t&1] and writes buf[1-(t&1)].
__global__ __launch_bounds__(256, 1) void
fused_ema_gru_fc(const float* __restrict__ x,
                 const float* __restrict__ W_ih,
                 const float* __restrict__ W_hh,
                 const float* __restrict__ b_ih,
                 const float* __restrict__ b_hh,
                 const float* __restrict__ W_fc,
                 const float* __restrict__ b_fc,
                 float* __restrict__ out)
{
    // h tile (16x128 bf16) stored pre-packed in WMMA A-fragment lane layout:
    // fragment s covers K=32s..32s+31; per lane L: rows M=L%16,
    // slots 0..7 -> K {0..7}+8*(L>=16), slots 8..15 -> K {16..23}+8*(L>=16).
    __shared__ __align__(32) __bf16 hfrag[2][4][512];
    __shared__ __align__(32) __bf16 xfrag[2][512];   // 16x32 A-frag, K=16..31 zero
    __shared__ float hT[16][HH];
    __shared__ float st1b[2][16];
    __shared__ float st2b[2][16];

    const int tid  = threadIdx.x;
    const int lane = tid & 31;
    const int w    = tid >> 5;            // wave id 0..7
    const int nlo  = lane & 15;
    const int b0   = blockIdx.x * 16;

    // ---- zero LDS A-fragments (h0 = 0, x-frag K padding = 0), both buffers ----
    #pragma unroll
    for (int i = 0; i < 16; ++i)
        ((__bf16*)hfrag)[tid + 256 * i] = (__bf16)0.0f;
    #pragma unroll
    for (int i = 0; i < 4; ++i)
        ((__bf16*)xfrag)[tid + 256 * i] = (__bf16)0.0f;

    // ---- gather weight B-fragments into registers (resident all 2048 steps) ----
    // B element (k,n) = W[n][k] since gemm is  h @ W^T.
    v16bf Bh[3][4];   // [gate r/z/n][k-slice]
    v16bf Bgi[3];     // gi weights, K padded 16->32 with zeros
    #pragma unroll
    for (int g = 0; g < 3; ++g) {
        const int n = 16 * (8 * g + w) + nlo;
        #pragma unroll
        for (int s = 0; s < 4; ++s) {
            v16bf b;
            #pragma unroll
            for (int idx = 0; idx < 16; ++idx) {
                const int kl = idx + ((idx >= 8) ? 8 : 0) + ((lane >= 16) ? 8 : 0);
                b[idx] = (__bf16)W_hh[n * HH + 32 * s + kl];
            }
            Bh[g][s] = b;
        }
        v16bf bi;
        #pragma unroll
        for (int idx = 0; idx < 16; ++idx) {
            const int kl = idx + ((idx >= 8) ? 8 : 0) + ((lane >= 16) ? 8 : 0);
            bi[idx] = (kl < FF) ? (__bf16)W_ih[n * FF + kl] : (__bf16)0.0f;
        }
        Bgi[g] = bi;
    }

    // biases for this wave's 16 hidden columns
    const int col = 16 * w + nlo;
    const float bias_r = b_ih[col]          + b_hh[col];
    const float bias_z = b_ih[HH + col]     + b_hh[HH + col];
    const float bih_n  = b_ih[2 * HH + col];
    const float bhh_n  = b_hh[2 * HH + col];

    // h-fragment store mapping (C/D layout -> A-frag layout) for this wave's col
    const int s_out  = col >> 5;
    const int klocal = col & 31;
    const int tl_off = ((klocal >> 3) & 1) * 16;
    const int slot   = (klocal & 7) + ((klocal >> 4) << 3);
    const int mb     = (lane < 16) ? 0 : 8;   // C layout: M = mb + vgpr_idx

    // EMA: one thread per (batch row, feature), state lives in registers
    const int erow = tid >> 4;
    const int ecol = tid & 15;
    const float* xptr = x + (size_t)(b0 + erow) * (size_t)(TT * FF) + ecol;
    const int xf_idx = (erow + ((ecol >= 8) ? 16 : 0)) * 16 + (ecol & 7);

    float e1 = xptr[0];              // EMA state for step 0 (x2[:,0] = x[:,0])
    float e2 = e1;
    xfrag[0][xf_idx] = (__bf16)e2;   // step-0 input tile into buffer 0

    const v8f zacc = (v8f)0.0f;
    v8f hprev = zacc;

    lds_drain_();      // prologue zeroing + x-frag(t=0) committed
    bar_signal_();     // matches the bar_wait_ at the top of iteration 0

    for (int t = 0; t < TT; ++t) {
        const int rp = t & 1;        // read buffer; write buffer is 1-rp

        // issue next-step global x load before the barrier wait: its latency
        // hides under the ds-loads + WMMA chain of this whole iteration
        const int tn = (t + 1 < TT) ? (t + 1) : t;
        const float xv = xptr[(size_t)tn * FF];

        bar_wait_();   // previous iteration's frag writes now visible

        // -- load A fragments (ds_load_b128 pairs, conflict-free lane layout) --
        const v16bf ax = *(const v16bf*)&xfrag[rp][lane * 16];
        v16bf ah[4];
        #pragma unroll
        for (int s = 0; s < 4; ++s)
            ah[s] = *(const v16bf*)&hfrag[rp][s][lane * 16];

        // -- 15 WMMAs: 3 gi (K=16 padded) + 12 gh (K=128) --
        v8f accR  = __builtin_amdgcn_wmma_f32_16x16x32_bf16(false, ax, false, Bgi[0], (short)0, zacc, false, false);
        v8f accZ  = __builtin_amdgcn_wmma_f32_16x16x32_bf16(false, ax, false, Bgi[1], (short)0, zacc, false, false);
        v8f accNi = __builtin_amdgcn_wmma_f32_16x16x32_bf16(false, ax, false, Bgi[2], (short)0, zacc, false, false);
        v8f accNh = zacc;
        #pragma unroll
        for (int s = 0; s < 4; ++s) {
            accR  = __builtin_amdgcn_wmma_f32_16x16x32_bf16(false, ah[s], false, Bh[0][s], (short)0, accR,  false, false);
            accZ  = __builtin_amdgcn_wmma_f32_16x16x32_bf16(false, ah[s], false, Bh[1][s], (short)0, accZ,  false, false);
            accNh = __builtin_amdgcn_wmma_f32_16x16x32_bf16(false, ah[s], false, Bh[2][s], (short)0, accNh, false, false);
        }

        // -- gates (all in-wave; C layout rows M = mb+vr, col fixed per lane) --
        v8f hnew = zacc;
        #pragma unroll
        for (int vr = 0; vr < 8; ++vr) {
            const float r = fsigmoid_(accR[vr] + bias_r);
            const float z = fsigmoid_(accZ[vr] + bias_z);
            const float n = ftanh_(accNi[vr] + bih_n + r * (accNh[vr] + bhh_n));
            hnew[vr] = n + z * (hprev[vr] - n);   // (1-z)*n + z*h
        }
        hprev = hnew;

        // -- EMA advance to step t+1 (uses the early global load) --
        const float n1 = xv + ALPHA * (e1 - xv);  // (1-a)*x + a*e1
        const float n2 = n1 + BETA  * (e2 - n1);  // (1-b)*e1' + b*e2
        const bool adv = (t + 1 < TT);
        e1 = adv ? n1 : e1;
        e2 = adv ? n2 : e2;

        if (ecol == 0)  // pull the x stream a cacheline ahead -> global_prefetch_b8
            __builtin_prefetch(&xptr[(size_t)(t + 16) * FF], 0, 1);

        // -- write next-step fragments into the OTHER buffer (no read hazard) --
        #pragma unroll
        for (int vr = 0; vr < 8; ++vr)
            hfrag[1 - rp][s_out][(mb + vr + tl_off) * 16 + slot] = (__bf16)hnew[vr];
        xfrag[1 - rp][xf_idx] = (__bf16)e2;

        lds_drain_();      // frag stores committed to LDS
        bar_signal_();     // single barrier per timestep; wait is at next top
    }
    bar_wait_();           // consume the final loop signal (keep barrier balanced)

    // ---- epilogue: out = hT @ W_fc^T + b_fc, then EMA-state corrections ----
    #pragma unroll
    for (int vr = 0; vr < 8; ++vr)
        hT[mb + vr][col] = hprev[vr];
    if (ecol == 1 || ecol == 2) {    // reference uses cols [1,2]
        st1b[ecol - 1][erow] = e1;
        st2b[ecol - 1][erow] = e2;
    }
    __syncthreads();

    if (tid < 32) {
        const int m = tid >> 1;
        const int c = tid & 1;
        float acc = b_fc[c];
        #pragma unroll 8
        for (int k = 0; k < HH; ++k)
            acc += hT[m][k] * W_fc[c * HH + k];
        acc = (acc - BETA  * st2b[c][m]) * (1.0f / (1.0f - BETA));
        acc = (acc - ALPHA * st1b[c][m]) * (1.0f / (1.0f - ALPHA));
        out[(size_t)(b0 + m) * CC + c] = acc;
    }
}

extern "C" void kernel_launch(void* const* d_in, const int* in_sizes, int n_in,
                              void* d_out, int out_size, void* d_ws, size_t ws_size,
                              hipStream_t stream) {
    (void)in_sizes; (void)n_in; (void)out_size; (void)d_ws; (void)ws_size;
    const float* x    = (const float*)d_in[0];
    const float* W_ih = (const float*)d_in[1];
    const float* W_hh = (const float*)d_in[2];
    const float* b_ih = (const float*)d_in[3];
    const float* b_hh = (const float*)d_in[4];
    const float* W_fc = (const float*)d_in[5];
    const float* b_fc = (const float*)d_in[6];
    float* out = (float*)d_out;

    dim3 grid(BB / 16);   // 64 WGs, one 16-row batch tile each
    dim3 block(256);      // 8 waves (wave32)
    hipLaunchKernelGGL(fused_ema_gru_fc, grid, block, 0, stream,
                       x, W_ih, W_hh, b_ih, b_hh, W_fc, b_fc, out);
}
